// FixedGemmaDecoder_42142219108651
// MI455X (gfx1250) — matrix operand order repaired and checked by reading, not compiled, not run on hardware
//
#include <hip/hip_runtime.h>
#include <hip/hip_bf16.h>
#include <math.h>
#include <stdint.h>

typedef __attribute__((ext_vector_type(16))) __bf16 v16bf;
typedef __attribute__((ext_vector_type(8)))  float  v8f;

#define SEQ 2048
#define HID 2048
#define AK  1024
#define LDSS 56   // LDS row stride in elements: 112 B = 16B-aligned, low-conflict

#if defined(__AMDGCN__) && __has_builtin(__builtin_amdgcn_tensor_load_to_lds) && \
    __has_builtin(__builtin_amdgcn_s_wait_tensorcnt)
#define USE_TDM 1
#else
#define USE_TDM 0
#endif

#if USE_TDM
typedef unsigned int tdm_v4u __attribute__((ext_vector_type(4)));
typedef int          tdm_v8i __attribute__((ext_vector_type(8)));
typedef int          tdm_v4i __attribute__((ext_vector_type(4)));

// Issue a TDM load of a 128-row x 32-element bf16 tile (row stride ld_elems in
// global) into LDS at lds_byte_addr, with 48 B of hardware row padding so rows
// land at the kernel's 112 B LDS stride.  D# per CDNA5 ISA ch.8.
__device__ __forceinline__
void tdm_tile_load_128x32(const __bf16* gsrc, unsigned lds_byte_addr, int ld_elems)
{
  unsigned long long ga = (unsigned long long)(uintptr_t)gsrc;
  unsigned ld = (unsigned)ld_elems;

  tdm_v4u g0;
  g0[0] = 1u;                                            // count=1, user descriptor
  g0[1] = lds_byte_addr;                                 // lds_addr
  g0[2] = (unsigned)ga;                                  // global_addr[31:0]
  g0[3] = (unsigned)((ga >> 32) & 0x01ffffffull)         // global_addr[56:32]
        | (2u << 30);                                    // type=2 ("image")

  tdm_v8i g1;
  g1[0] = (int)((1u << 16)        // data_size = 2 bytes
              | (1u << 20)        // pad_enable
              | (3u << 22)        // pad_interval: 16 dwords (= one 64 B row)
              | (11u << 25));     // pad_amount: 12 dwords (= 48 B)
  g1[1] = (int)((ld & 0xffffu) << 16);                   // tensor_dim0[15:0]
  g1[2] = (int)(((ld >> 16) & 0xffffu) | (128u << 16));  // tensor_dim0[31:16], tensor_dim1=128
  g1[3] = (int)(32u << 16);                              // tile_dim0 = 32
  g1[4] = (int)128;                                      // tile_dim1 = 128 (tile_dim2=0)
  g1[5] = (int)ld;                                       // tensor_dim0_stride[31:0]
  g1[6] = 0;
  g1[7] = 0;

  tdm_v4i z4 = {0, 0, 0, 0};
#if __clang_major__ >= 23
  tdm_v8i z8 = {0, 0, 0, 0, 0, 0, 0, 0};
  __builtin_amdgcn_tensor_load_to_lds(g0, g1, z4, z4, z8, 0);
#else
  __builtin_amdgcn_tensor_load_to_lds(g0, g1, z4, z4, 0);
#endif
}
#endif // USE_TDM

// ---------------------------------------------------------------------------
// fp32 -> bf16 conversion (reference rounds all GEMM operands to bf16)
// ---------------------------------------------------------------------------
__global__ __launch_bounds__(256)
void cvt_f32_bf16(const float* __restrict__ in, __bf16* __restrict__ out, int n) {
  int i = blockIdx.x * 256 + threadIdx.x;
  if (i < n) out[i] = (__bf16)in[i];
}

// ---------------------------------------------------------------------------
// Tiled bf16 GEMM on WMMA: C[M][N] = bf16( A[M][K] x B ), fp32 accumulation.
//   BT=false: B is row-major [K][ldb], element B[k*ldb + n]
//   BT=true : B is row-major [N][ldb], element B[n*ldb + k]  (i.e. C = A x B^T)
//   EPI==0  : C = bf16(acc)
//   EPI==1  : C = bf16( f32(bf16(acc)) + f32(resid) )   (residual add)
// Block tile 128x128, BK=32. 256 threads = 8 waves; wave grid 4(M) x 2(N),
// each wave computes 32x64 via a 2x4 grid of v_wmma_f32_16x16x32_bf16.
// Software pipeline: TDM (or register-staged) loads for tile k+1 overlap the
// WMMA work of tile k; LDS ping-pong -> single barrier per K-step.
// A-tile (and B-tile when BT) staged by the Tensor Data Mover; the transposed
// B path stays on register staging since TDM does not transpose.
// ---------------------------------------------------------------------------
template <int EPI, bool BT>
__global__ __launch_bounds__(256)
void gemm_bf16_wmma(const __bf16* __restrict__ A, const __bf16* __restrict__ B,
                    __bf16* __restrict__ C, const __bf16* __restrict__ resid,
                    int M, int N, int K, int lda, int ldb, int ldc)
{
  __shared__ __align__(16) __bf16 sA[2][128][LDSS];
  __shared__ __align__(16) __bf16 sB[2][128][LDSS];   // stored as [n][k]

  const int tid  = threadIdx.x;
  const int lane = tid & 31;
  const int wave = tid >> 5;
  const int wm   = (wave & 3) * 32;   // wave's M offset inside block tile
  const int wn   = (wave >> 2) * 64;  // wave's N offset inside block tile
  const int bm   = blockIdx.y * 128;
  const int bn   = blockIdx.x * 128;

  v8f acc[2][4] = {};
  union Frag { v16bf v; uint4 q[2]; };

  uint4 ra[2], rb[2];   // register staging (fallback / transposed-B path)

  auto load_a_regs = [&](int kt) {
#pragma unroll
    for (int i = 0; i < 2; ++i) {
      int c   = tid + i * 256;
      int row = c >> 2;
      int col = (c & 3) * 8;
      ra[i] = *reinterpret_cast<const uint4*>(&A[(size_t)(bm + row) * lda + kt + col]);
    }
  };
  auto store_a_lds = [&](int buf) {
#pragma unroll
    for (int i = 0; i < 2; ++i) {
      int c   = tid + i * 256;
      int row = c >> 2;
      int col = (c & 3) * 8;
      *reinterpret_cast<uint4*>(&sA[buf][row][col]) = ra[i];
    }
  };
  auto load_b_regs = [&](int kt) {
    if (BT) {
#pragma unroll
      for (int i = 0; i < 2; ++i) {
        int c   = tid + i * 256;
        int row = c >> 2;
        int col = (c & 3) * 8;
        rb[i] = *reinterpret_cast<const uint4*>(&B[(size_t)(bn + row) * ldb + kt + col]);
      }
    } else {
#pragma unroll
      for (int i = 0; i < 2; ++i) {
        int c    = tid + i * 256;
        int krow = c >> 4;
        int col  = (c & 15) * 8;
        rb[i] = *reinterpret_cast<const uint4*>(&B[(size_t)(kt + krow) * ldb + bn + col]);
      }
    }
  };
  auto store_b_lds = [&](int buf) {
    if (BT) {
#pragma unroll
      for (int i = 0; i < 2; ++i) {
        int c   = tid + i * 256;
        int row = c >> 2;
        int col = (c & 3) * 8;
        *reinterpret_cast<uint4*>(&sB[buf][row][col]) = rb[i];
      }
    } else {
#pragma unroll
      for (int i = 0; i < 2; ++i) {
        int c    = tid + i * 256;
        int krow = c >> 4;
        int col  = (c & 15) * 8;
        union { uint4 q; __bf16 h[8]; } d;
        d.q = rb[i];
#pragma unroll
        for (int e = 0; e < 8; ++e) sB[buf][col + e][krow] = d.h[e];
      }
    }
  };

#if USE_TDM
  const unsigned ldsA[2] = { (unsigned)(uintptr_t)&sA[0][0][0],
                             (unsigned)(uintptr_t)&sA[1][0][0] };
  const unsigned ldsB[2] = { (unsigned)(uintptr_t)&sB[0][0][0],
                             (unsigned)(uintptr_t)&sB[1][0][0] };
  // ---- prologue: stage tile 0 ----
  if (wave == 0) {
    tdm_tile_load_128x32(&A[(size_t)bm * lda], ldsA[0], lda);
    if (BT) tdm_tile_load_128x32(&B[(size_t)bn * ldb], ldsB[0], ldb);
  }
  if (!BT) { load_b_regs(0); store_b_lds(0); }
  if (wave == 0) __builtin_amdgcn_s_wait_tensorcnt(0);
#else
  load_a_regs(0); load_b_regs(0);
  store_a_lds(0); store_b_lds(0);
#endif
  __syncthreads();

  int cur = 0;
  for (int kt = 0; kt < K; kt += 32) {
    const bool has_next = (kt + 32 < K);
    if (has_next) {
#if USE_TDM
      if (wave == 0) {            // DMA next tile while this wave's WMMAs run
        tdm_tile_load_128x32(&A[(size_t)bm * lda + kt + 32], ldsA[cur ^ 1], lda);
        if (BT) tdm_tile_load_128x32(&B[(size_t)bn * ldb + kt + 32], ldsB[cur ^ 1], ldb);
      }
      if (!BT) load_b_regs(kt + 32);
#else
      load_a_regs(kt + 32);
      load_b_regs(kt + 32);
#endif
      if (!BT && kt + 64 < K)     // keep L2 warm two tiles ahead for the VMEM path
        __builtin_prefetch(&B[(size_t)(kt + 64 + (tid >> 4)) * ldb + bn + (tid & 15) * 8], 0, 1);
    }

    // ---- load fragments (ISA 7.12.2 layouts) and issue 8 WMMAs ----
    Frag fa[2], fb[4];
#pragma unroll
    for (int mt = 0; mt < 2; ++mt) {
      int row = wm + mt * 16 + (lane & 15);
      int kb  = (lane >> 4) * 8;          // lanes 0-15: K 0..7/16..23, lanes 16-31: 8..15/24..31
      fa[mt].q[0] = *reinterpret_cast<const uint4*>(&sA[cur][row][kb]);
      fa[mt].q[1] = *reinterpret_cast<const uint4*>(&sA[cur][row][kb + 16]);
    }
#pragma unroll
    for (int nt = 0; nt < 4; ++nt) {
      int n  = wn + nt * 16 + (lane & 15);
      int kb = (lane >> 4) * 8;           // lanes 0-15: K 0..7/16..23, lanes 16-31: 8..15/24..31
      fb[nt].q[0] = *reinterpret_cast<const uint4*>(&sB[cur][n][kb]);
      fb[nt].q[1] = *reinterpret_cast<const uint4*>(&sB[cur][n][kb + 16]);
    }
#pragma unroll
    for (int mt = 0; mt < 2; ++mt)
#pragma unroll
      for (int nt = 0; nt < 4; ++nt)
        acc[mt][nt] = __builtin_amdgcn_wmma_f32_16x16x32_bf16(
            false, fa[mt].v, false, fb[nt].v, (short)0, acc[mt][nt], false, false);

    if (has_next) {
#if USE_TDM
      if (!BT) store_b_lds(cur ^ 1);
      if (wave == 0) __builtin_amdgcn_s_wait_tensorcnt(0);
#else
      store_a_lds(cur ^ 1);
      store_b_lds(cur ^ 1);
#endif
      __syncthreads();                    // single barrier per K-step
      cur ^= 1;
    }
  }

  // ---- epilogue: C/D layout -> row = 8*(lane>=16)+r, col = lane&15 ----
#pragma unroll
  for (int mt = 0; mt < 2; ++mt) {
#pragma unroll
    for (int nt = 0; nt < 4; ++nt) {
      int col = bn + wn + nt * 16 + (lane & 15);
#pragma unroll
      for (int r = 0; r < 8; ++r) {
        int row = bm + wm + mt * 16 + (lane >> 4) * 8 + r;
        float f  = acc[mt][nt][r];
        __bf16 h = (__bf16)f;                          // bf16_matmul result rounding
        if constexpr (EPI == 1) {
          float s = (float)h + (float)resid[(size_t)row * ldc + col];
          h = (__bf16)s;                               // residual re-round
        }
        C[(size_t)row * ldc + col] = h;
      }
    }
  }
}

// ---------------------------------------------------------------------------
// Row softmax: x = bf16( f32(score)*scale ); fp32 stable softmax; bf16 out.
// One 256-thread block per row of 1024 columns.
// ---------------------------------------------------------------------------
__global__ __launch_bounds__(256)
void softmax_rows(const __bf16* __restrict__ x, __bf16* __restrict__ y,
                  int ncols, float scale)
{
  __shared__ float red[256];
  const int row = blockIdx.x;
  const __bf16* xr = x + (size_t)row * ncols;
  __bf16*       yr = y + (size_t)row * ncols;

  float v[4];
  float m = -INFINITY;
#pragma unroll
  for (int i = 0; i < 4; ++i) {
    int c = threadIdx.x + i * 256;
    float s = (float)xr[c] * scale;
    float xb = (float)(__bf16)s;            // reference rounds scaled scores to bf16
    v[i] = xb;
    m = fmaxf(m, xb);
  }
  red[threadIdx.x] = m; __syncthreads();
  for (int s = 128; s > 0; s >>= 1) {
    if (threadIdx.x < s) red[threadIdx.x] = fmaxf(red[threadIdx.x], red[threadIdx.x + s]);
    __syncthreads();
  }
  m = red[0]; __syncthreads();

  float sum = 0.f;
#pragma unroll
  for (int i = 0; i < 4; ++i) { v[i] = expf(v[i] - m); sum += v[i]; }
  red[threadIdx.x] = sum; __syncthreads();
  for (int s = 128; s > 0; s >>= 1) {
    if (threadIdx.x < s) red[threadIdx.x] += red[threadIdx.x + s];
    __syncthreads();
  }
  float inv = 1.0f / red[0];
#pragma unroll
  for (int i = 0; i < 4; ++i)
    yr[threadIdx.x + i * 256] = (__bf16)(v[i] * inv);
}

// ---------------------------------------------------------------------------
// GeGLU: gate <- bf16( f32(bf16(gelu_tanh(f32(gate)))) * f32(up) )
// ---------------------------------------------------------------------------
__global__ __launch_bounds__(256)
void geglu(__bf16* __restrict__ gate, const __bf16* __restrict__ up, int n) {
  int i = blockIdx.x * 256 + threadIdx.x;
  if (i < n) {
    float g = (float)gate[i];
    float a = 0.5f * g * (1.0f + tanhf(0.7978845608028654f * (g + 0.044715f * g * g * g)));
    float af = (float)(__bf16)a;            // activated rounded to bf16 first
    gate[i] = (__bf16)(af * (float)up[i]);
  }
}

// ---------------------------------------------------------------------------
extern "C" void kernel_launch(void* const* d_in, const int* in_sizes, int n_in,
                              void* d_out, int out_size, void* d_ws, size_t ws_size,
                              hipStream_t stream)
{
  const float* hidden_f = (const float*)d_in[0];
  const float* w_f[7] = { (const float*)d_in[1], (const float*)d_in[2],
                          (const float*)d_in[3], (const float*)d_in[4],
                          (const float*)d_in[5], (const float*)d_in[6],
                          (const float*)d_in[7] };

  const size_t H2 = (size_t)HID * HID;           // 4M elements
  const size_t SA = (size_t)SEQ * AK;            // 2M elements

  __bf16* p   = (__bf16*)d_ws;
  __bf16* hB  = p; p += H2;       // hidden bf16
  __bf16* w[7];
  for (int i = 0; i < 7; ++i) { w[i] = p; p += H2; }   // q,k,v,o,gate,up,down
  __bf16* q      = p; p += H2;
  __bf16* k      = p; p += H2;
  __bf16* v      = p; p += H2;
  __bf16* scores = p; p += SA;
  __bf16* probs  = p; p += SA;
  __bf16* ctx    = p; p += H2;
  __bf16* post   = p; p += H2;
  __bf16* gate   = p; p += H2;
  __bf16* up     = p; p += H2;

  const int CVB = (int)(H2 / 256);
  cvt_f32_bf16<<<CVB, 256, 0, stream>>>(hidden_f, hB, (int)H2);
  for (int i = 0; i < 7; ++i)
    cvt_f32_bf16<<<CVB, 256, 0, stream>>>(w_f[i], w[i], (int)H2);

  dim3 blk(256);
  dim3 gSq(HID / 128, SEQ / 128);                 // 16 x 16
  dim3 gSc(AK / 128, SEQ / 128);                  // 8 x 16

  // q/k/v projections
  gemm_bf16_wmma<0, false><<<gSq, blk, 0, stream>>>(hB, w[0], q, nullptr,
      SEQ, HID, HID, HID, HID, HID);
  gemm_bf16_wmma<0, false><<<gSq, blk, 0, stream>>>(hB, w[1], k, nullptr,
      SEQ, HID, HID, HID, HID, HID);
  gemm_bf16_wmma<0, false><<<gSq, blk, 0, stream>>>(hB, w[2], v, nullptr,
      SEQ, HID, HID, HID, HID, HID);

  // scores = q x k[:AK]^T   (B transposed: k rows are the N dimension)
  gemm_bf16_wmma<0, true><<<gSc, blk, 0, stream>>>(q, k, scores, nullptr,
      SEQ, AK, HID, HID, HID, AK);

  // softmax(scores / sqrt(HID))
  softmax_rows<<<SEQ, blk, 0, stream>>>(scores, probs, AK, 0.022097086912079610f);

  // context = probs x v[:AK]
  gemm_bf16_wmma<0, false><<<gSq, blk, 0, stream>>>(probs, v, ctx, nullptr,
      SEQ, HID, AK, AK, HID, HID);

  // post_attention = bf16(hidden + bf16(ctx x o_proj))   (fused residual)
  gemm_bf16_wmma<1, false><<<gSq, blk, 0, stream>>>(ctx, w[3], post, hB,
      SEQ, HID, HID, HID, HID, HID);

  // GeGLU MLP
  gemm_bf16_wmma<0, false><<<gSq, blk, 0, stream>>>(post, w[4], gate, nullptr,
      SEQ, HID, HID, HID, HID, HID);
  gemm_bf16_wmma<0, false><<<gSq, blk, 0, stream>>>(post, w[5], up, nullptr,
      SEQ, HID, HID, HID, HID, HID);
  geglu<<<CVB, blk, 0, stream>>>(gate, up, (int)H2);

  // out = bf16(post + bf16(gated x down_proj))   (fused residual, direct to d_out)
  gemm_bf16_wmma<1, false><<<gSq, blk, 0, stream>>>(gate, w[6], (__bf16*)d_out, post,
      SEQ, HID, HID, HID, HID, HID);
}
